// LiquidLayer_61538291417814
// MI455X (gfx1250) — matrix operand order repaired and checked by reading, not compiled, not run on hardware
//
#include <hip/hip_runtime.h>

// ---------------------------------------------------------------------------
// LTC (liquid time-constant) forward for MI455X / gfx1250, wave32 + WMMA.
//   B=32, T=512, I=256, H=512, ODE=4.
// ---------------------------------------------------------------------------

#define BATCH 32
#define SEQ   512
#define INSZ  256
#define HID   512
#define ODE   4
#define NWG   16           // persistent workgroups for the scan (H/32 slices)

typedef __attribute__((ext_vector_type(8)))  float  v8f;
typedef __attribute__((ext_vector_type(2)))  float  v2f;
typedef __attribute__((ext_vector_type(16))) __bf16 v16bf;

#if defined(__has_builtin)
#if __has_builtin(__builtin_amdgcn_wmma_f32_16x16x4_f32)
#define HAVE_WMMA_F32 1
#endif
#endif

union BfFrag { v16bf v; uint4 q[2]; };

// ---------------------------------------------------------------------------
// Kernel 1: pack W_rec and W_tau[:,256:] to bf16, zero h ping-pong + barrier.
// ---------------------------------------------------------------------------
__global__ __launch_bounds__(256) void ltc_pack_kernel(
    const float* __restrict__ W_rec, const float* __restrict__ W_tau,
    __bf16* __restrict__ wrec_bf, __bf16* __restrict__ wtauh_bf,
    __bf16* __restrict__ hb0, __bf16* __restrict__ hb1,
    unsigned* __restrict__ bar)
{
  int idx = blockIdx.x * 256 + threadIdx.x;          // 0 .. 512*512-1
  if (idx < HID * HID) {
    int n = idx >> 9, k = idx & 511;
    wrec_bf[idx]  = (__bf16)W_rec[idx];
    wtauh_bf[idx] = (__bf16)W_tau[(size_t)n * (INSZ + HID) + INSZ + k];
    if (idx < BATCH * HID) { hb0[idx] = (__bf16)0.0f; hb1[idx] = (__bf16)0.0f; }
    if (idx == 0) *bar = 0u;
  }
}

// ---------------------------------------------------------------------------
// Kernel 2: input projections with full-precision f32 WMMA (16x16x4).
//   mat0: xin [t][b][n] = x @ W_in.T + b_in
//   mat1: xtau[t][b][n] = x @ W_tau_x.T + b_tau
//   mat2: y   [b][t][n] = x @ W_res.T      (residual, straight into d_out)
// ---------------------------------------------------------------------------
__global__ __launch_bounds__(256) void ltc_proj_kernel(
    const float* __restrict__ x,
    const float* __restrict__ W_in,  const float* __restrict__ b_in,
    const float* __restrict__ W_tau, const float* __restrict__ b_tau,
    const float* __restrict__ W_res,
    float* __restrict__ xin, float* __restrict__ xtau, float* __restrict__ y)
{
  const int l  = threadIdx.x & 31;
  const int w  = threadIdx.x >> 5;
  const int lm = l & 15, lh = l >> 4;
  const int wave   = blockIdx.x * 8 + w;
  const int stride = gridDim.x * 8;
  const int MT = (BATCH * SEQ) / 16;                 // 1024 M-tiles
  const int NT = HID / 16;                           // 32 N-tiles
  const int JOBS = MT * NT * 3;

  for (int j = wave; j < JOBS; j += stride) {
    const int mat   = j % 3;
    const int tile  = j / 3;
    const int ntile = tile & (NT - 1);
    const int mtile = tile / NT;

    const float* W;  int ldw;  const float* bias;  float* outp;
    if (mat == 0)      { W = W_in;  ldw = INSZ;        bias = b_in;  outp = xin;  }
    else if (mat == 1) { W = W_tau; ldw = INSZ + HID;  bias = b_tau; outp = xtau; }
    else               { W = W_res; ldw = INSZ;        bias = nullptr; outp = y;  }

    const int col = ntile * 16 + lm;
    const float* xrow = x + (size_t)(mtile * 16 + lm) * INSZ;
    const float* wrow = W + (size_t)col * ldw;

    v8f c = {};
#if HAVE_WMMA_F32
    #pragma unroll 4
    for (int k0 = 0; k0 < INSZ; k0 += 4) {
      v2f a = *(const v2f*)(xrow + k0 + lh * 2);
      v2f b = *(const v2f*)(wrow + k0 + lh * 2);
      c = __builtin_amdgcn_wmma_f32_16x16x4_f32(false, a, false, b,
                                                (short)0, c, false, false);
    }
#else
    #pragma unroll
    for (int r = 0; r < 8; ++r) {
      const float* xr = x + (size_t)(mtile * 16 + r + 8 * lh) * INSZ;
      float s = 0.0f;
      for (int k = 0; k < INSZ; ++k) s += xr[k] * wrow[k];
      c[r] = s;
    }
#endif
    const float bv = bias ? bias[col] : 0.0f;
    #pragma unroll
    for (int r = 0; r < 8; ++r) {
      int mg = mtile * 16 + r + 8 * lh;              // flattened (b, t)
      int b_ = mg >> 9, t_ = mg & 511;
      float val = c[r] + bv;
      size_t idx = (mat == 2) ? ((size_t)(b_ * SEQ + t_) * HID + col)   // [b][t][n]
                              : ((size_t)(t_ * BATCH + b_) * HID + col);// [t][b][n]
      outp[idx] = val;
    }
  }
}

// ---------------------------------------------------------------------------
// Grid-wide split barrier (agent-scope atomics, L2-resident counter).
// ---------------------------------------------------------------------------
__device__ __forceinline__ void grid_sync(unsigned* bar, unsigned* ep)
{
  __syncthreads();
  if (threadIdx.x == 0) {
    __threadfence();
    __hip_atomic_fetch_add(bar, 1u, __ATOMIC_RELEASE, __HIP_MEMORY_SCOPE_AGENT);
    *ep += NWG;
    while (__hip_atomic_load(bar, __ATOMIC_ACQUIRE, __HIP_MEMORY_SCOPE_AGENT) < *ep)
      __builtin_amdgcn_s_sleep(2);
  }
  __syncthreads();
}

// ---------------------------------------------------------------------------
// Kernel 3: persistent recurrent scan.
//   16 WGs x 128 threads (4 waves). WG 'g' owns hidden slice [32g, 32g+32).
//   Wave = (mtile in batch, ntile in slice) -> one 16x16 h-tile in registers
//   (exactly the WMMA C layout). Per ODE step: K=512 reduction with
//   v_wmma_f32_16x16x32_bf16 against LDS-resident bf16 weight slices,
//   shared A-fragment (h in bf16 ping-pong, L2-resident), then
//   sigmoid/tanh blend in f32 registers, publish, grid barrier.
// ---------------------------------------------------------------------------
__global__ __launch_bounds__(128, 1) void ltc_scan_kernel(
    const float* __restrict__ xin_all, const float* __restrict__ xtau_all,
    const __bf16* __restrict__ wrec_bf, const __bf16* __restrict__ wtauh_bf,
    __bf16* __restrict__ hb0, __bf16* __restrict__ hb1,
    float* __restrict__ y, float* __restrict__ hfin,
    unsigned* __restrict__ bar)
{
  __shared__ __bf16 Ws[2][32][HID];                  // 64 KB: W_rec | W_tau_h slices

  const int tid = threadIdx.x;
  const int n0  = blockIdx.x * 32;

  // Stage this WG's weight slices into LDS (16B chunks).
  for (int i = tid; i < 32 * 64; i += 128) {
    int row = i >> 6, seg = (i & 63) * 8;
    *(uint4*)(&Ws[0][row][seg]) = *(const uint4*)(wrec_bf  + (size_t)(n0 + row) * HID + seg);
    *(uint4*)(&Ws[1][row][seg]) = *(const uint4*)(wtauh_bf + (size_t)(n0 + row) * HID + seg);
  }
  __syncthreads();

  const int l = tid & 31, w = tid >> 5;
  const int lm = l & 15, lh = l >> 4;
  const int mtile = w & 1, ntile = w >> 1;
  const int rowA = mtile * 16 + lm;                  // batch row for A fragment
  const int nl   = ntile * 16 + lm;                  // col within slice
  const int gcol = n0 + nl;                          // global hidden col

  v8f h_acc = {};
  unsigned ep = 0;
  int p = 0;

  for (int t = 0; t < SEQ; ++t) {
    const float* xin_t  = xin_all  + (size_t)t * BATCH * HID;
    const float* xtau_t = xtau_all + (size_t)t * BATCH * HID;
    if (t + 1 < SEQ) {                               // CDNA5 global_prefetch_b8
      __builtin_prefetch(xin_all  + (size_t)(t + 1) * BATCH * HID + tid * 32, 0, 1);
      __builtin_prefetch(xtau_all + (size_t)(t + 1) * BATCH * HID + tid * 32, 0, 1);
    }
    v8f xin_f, xtau_f;
    #pragma unroll
    for (int r = 0; r < 8; ++r) {
      int b = mtile * 16 + r + 8 * lh;
      xin_f[r]  = xin_t [(size_t)b * HID + gcol];
      xtau_f[r] = xtau_t[(size_t)b * HID + gcol];
    }

    for (int s = 0; s < ODE; ++s) {
      const __bf16* hr = p ? hb1 : hb0;
      __bf16*       hw = p ? hb0 : hb1;
      v8f accF = xin_f, accA = xtau_f;

      for (int k0 = 0; k0 < HID; k0 += 32) {
        BfFrag a, br, bt;
        const __bf16* ap = hr + (size_t)rowA * HID + k0 + 8 * lh;
        a.q[0] = *(const uint4*)(ap);
        a.q[1] = *(const uint4*)(ap + 16);
        const __bf16* wr = &Ws[0][nl][k0 + 8 * lh];
        const __bf16* wt = &Ws[1][nl][k0 + 8 * lh];
        br.q[0] = *(const uint4*)(wr);  br.q[1] = *(const uint4*)(wr + 16);
        bt.q[0] = *(const uint4*)(wt);  bt.q[1] = *(const uint4*)(wt + 16);
        accF = __builtin_amdgcn_wmma_f32_16x16x32_bf16(false, a.v, false, br.v,
                                                       (short)0, accF, false, false);
        accA = __builtin_amdgcn_wmma_f32_16x16x32_bf16(false, a.v, false, bt.v,
                                                       (short)0, accA, false, false);
      }

      #pragma unroll
      for (int r = 0; r < 8; ++r) {
        float alpha = 1.0f / (1.0f + __expf(-accA[r]));
        float f     = tanhf(accF[r]);
        h_acc[r]    = h_acc[r] + alpha * (f - h_acc[r]);   // (1-a)h + a f
      }
      #pragma unroll
      for (int r = 0; r < 8; ++r) {
        int b = mtile * 16 + r + 8 * lh;
        hw[(size_t)b * HID + gcol] = (__bf16)h_acc[r];
      }
      grid_sync(bar, &ep);
      p ^= 1;
    }

    #pragma unroll
    for (int r = 0; r < 8; ++r) {
      int b = mtile * 16 + r + 8 * lh;
      size_t yi = ((size_t)b * SEQ + t) * HID + gcol;
      y[yi] += h_acc[r];                              // residual already there
      if (t == SEQ - 1) hfin[(size_t)b * HID + gcol] = h_acc[r];
    }
  }
}

// ---------------------------------------------------------------------------
// Kernel 4: in-place LayerNorm over hidden dim (one block per (b,t) row).
// ---------------------------------------------------------------------------
__global__ __launch_bounds__(256) void ltc_ln_kernel(
    float* __restrict__ y, const float* __restrict__ gamma,
    const float* __restrict__ beta)
{
  const int tid = threadIdx.x;
  float* yr = y + (size_t)blockIdx.x * HID;
  float v0 = yr[tid], v1 = yr[tid + 256];
  __shared__ float s1[256], s2[256];
  s1[tid] = v0 + v1;
  s2[tid] = v0 * v0 + v1 * v1;
  __syncthreads();
  for (int off = 128; off > 0; off >>= 1) {
    if (tid < off) { s1[tid] += s1[tid + off]; s2[tid] += s2[tid + off]; }
    __syncthreads();
  }
  float mu   = s1[0] * (1.0f / HID);
  float var  = s2[0] * (1.0f / HID) - mu * mu;
  float rstd = rsqrtf(var + 1e-5f);
  yr[tid]       = (v0 - mu) * rstd * gamma[tid]       + beta[tid];
  yr[tid + 256] = (v1 - mu) * rstd * gamma[tid + 256] + beta[tid + 256];
}

// ---------------------------------------------------------------------------
extern "C" void kernel_launch(void* const* d_in, const int* in_sizes, int n_in,
                              void* d_out, int out_size, void* d_ws, size_t ws_size,
                              hipStream_t stream)
{
  (void)in_sizes; (void)n_in; (void)out_size; (void)ws_size;
  const float* x     = (const float*)d_in[0];
  const float* W_in  = (const float*)d_in[1];
  const float* b_in  = (const float*)d_in[2];
  const float* W_rec = (const float*)d_in[3];
  const float* W_tau = (const float*)d_in[4];
  const float* b_tau = (const float*)d_in[5];
  const float* W_res = (const float*)d_in[6];
  const float* gamma = (const float*)d_in[7];
  const float* beta  = (const float*)d_in[8];

  float* y    = (float*)d_out;                        // (B, T, H)
  float* hfin = y + (size_t)BATCH * SEQ * HID;        // (B, H) tail

  // Workspace layout (~65.1 MB).
  float*  xin      = (float*)d_ws;                    // [T][B][H] f32
  float*  xtau     = xin  + (size_t)SEQ * BATCH * HID;
  __bf16* wrec_bf  = (__bf16*)(xtau + (size_t)SEQ * BATCH * HID);
  __bf16* wtauh_bf = wrec_bf  + (size_t)HID * HID;
  __bf16* hb0      = wtauh_bf + (size_t)HID * HID;
  __bf16* hb1      = hb0 + (size_t)BATCH * HID;
  unsigned* bar    = (unsigned*)(hb1 + (size_t)BATCH * HID);

  ltc_pack_kernel<<<(HID * HID) / 256, 256, 0, stream>>>(
      W_rec, W_tau, wrec_bf, wtauh_bf, hb0, hb1, bar);
  ltc_proj_kernel<<<512, 256, 0, stream>>>(
      x, W_in, b_in, W_tau, b_tau, W_res, xin, xtau, y);
  ltc_scan_kernel<<<NWG, 128, 0, stream>>>(
      xin, xtau, wrec_bf, wtauh_bf, hb0, hb1, y, hfin, bar);
  ltc_ln_kernel<<<BATCH * SEQ, 256, 0, stream>>>(y, gamma, beta);
}
